// Attention_22737556865237
// MI455X (gfx1250) — matrix operand order repaired and checked by reading, not compiled, not run on hardware
//
#include <hip/hip_runtime.h>
#include <hip/hip_fp16.h>

typedef __attribute__((ext_vector_type(16))) _Float16 v16h;
typedef __attribute__((ext_vector_type(8)))  _Float16 v8h;
typedef __attribute__((ext_vector_type(8)))  float    v8f;
typedef __attribute__((ext_vector_type(4)))  unsigned int v4u;
typedef __attribute__((ext_vector_type(8)))  int v8i;
typedef __attribute__((ext_vector_type(4)))  int v4i;

#define B_   2
#define S_   2048
#define D_   768
#define H_   12
#define HD_  64
#define SCALE_ 0.125f  // HD^-0.5

// ---------------------------------------------------------------------------
// WMMA helper (CDNA5 v_wmma_f32_16x16x32_f16, wave32)
// ---------------------------------------------------------------------------
__device__ __forceinline__ v8f wmma16x16x32(v16h a, v16h b, v8f c) {
    return __builtin_amdgcn_wmma_f32_16x16x32_f16(false, a, false, b,
                                                  (short)0, c, false, false);
}

// A-fragment (16x32) from row-major storage (global or LDS), ld in elements.
// Lane (half=lane>>4, r=lane&15) holds row M=r; slot j -> K = k0 +
// (j<8 ? j : j+8) + 8*half  => two contiguous 16B loads.
__device__ __forceinline__ v16h load_a_frag(const _Float16* base, int ld,
                                            int row, int k0, int half) {
    const _Float16* p = base + (size_t)row * ld + k0 + 8 * half;
    v8h lo = *(const v8h*)p;
    v8h hi = *(const v8h*)(p + 16);
    return __builtin_shufflevector(lo, hi, 0,1,2,3,4,5,6,7,8,9,10,11,12,13,14,15);
}

// B-fragment (32x16) from an N-major matrix: element (K=k, N=n) at
// baseT[n*ld + k]. Lane holds column N=col; slot j -> K = k0 +
// (j<8 ? j : j+8) + 8*half.
__device__ __forceinline__ v16h load_b_frag(const _Float16* baseT, int ld,
                                            int col, int k0, int half) {
    const _Float16* p = baseT + (size_t)col * ld + k0 + 8 * half;
    v8h lo = *(const v8h*)p;
    v8h hi = *(const v8h*)(p + 16);
    return __builtin_shufflevector(lo, hi, 0,1,2,3,4,5,6,7,8,9,10,11,12,13,14,15);
}

// ---------------------------------------------------------------------------
// Tensor Data Mover: stage a (128 rows) x (32 contiguous f16) tile into LDS.
// 2D descriptor per ISA 8.3/8.4: group0 {count=1, lds_addr, global_addr,
// type=2}; group1 {data_size=2B, tensor dims, tile 32x128, dim0 stride}.
// Higher groups zero (<=2D tensor).  Tracked by TENSORcnt.
// ---------------------------------------------------------------------------
__device__ __forceinline__ uint32_t lds_off(const void* p) {
    return (uint32_t)(uintptr_t)p;   // LDS aperture: low 32 bits = byte offset
}

__device__ __forceinline__ void tdm_load_tile_128x32(const _Float16* gsrc,
                                                     uint32_t lds_byte_off,
                                                     int tensor_ld /*elems*/) {
    uint64_t ga = (uint64_t)(uintptr_t)gsrc;
    v4u g0;
    g0[0] = 1u;                                  // count=1, user descriptor
    g0[1] = lds_byte_off;                        // lds_addr (bytes)
    g0[2] = (uint32_t)ga;                        // global_addr[31:0]
    g0[3] = (uint32_t)((ga >> 32) & 0x1FFFFFFu)  // global_addr[56:32]
            | (2u << 30);                        // type=2 ("image")
    uint32_t dim0 = (uint32_t)tensor_ld;         // K extent (elements)
    uint32_t dim1 = 1u << 20;                    // large: no OOB clipping
    uint64_t s0   = (uint64_t)tensor_ld;         // dim0 stride (elements)
    v8i g1;
    g1[0] = (int)(1u << 16);                     // data_size=1 -> 2 bytes
    g1[1] = (int)((dim0 & 0xFFFFu) << 16);       // tensor_dim0[15:0]
    g1[2] = (int)((dim0 >> 16) | ((dim1 & 0xFFFFu) << 16));
    g1[3] = (int)((dim1 >> 16) | (32u << 16));   // tile_dim0 = 32
    g1[4] = (int)128;                            // tile_dim1 = 128, tile_dim2=0
    g1[5] = (int)(uint32_t)(s0 & 0xFFFFFFFFu);   // dim0_stride[31:0]
    g1[6] = (int)(uint32_t)(s0 >> 32);           // dim0_stride[47:32]
    g1[7] = 0;                                   // dim1_stride (unused for 2D)
    v4i z4 = {};
    v8i z8 = {};
    // 6-arg form on this toolchain: (g0, g1, g2, g3, extra, cpol)
    __builtin_amdgcn_tensor_load_to_lds(g0, g1, z4, z4, z8, 0);
}

// ---------------------------------------------------------------------------
// Kernel 1: fp32 -> f16 conversions (+ weight transposes to N-major)
// ---------------------------------------------------------------------------
__global__ void __launch_bounds__(256) convert_kernel(
    const float* __restrict__ x, const float* __restrict__ wqkv,
    const float* __restrict__ wproj,
    _Float16* __restrict__ x16, _Float16* __restrict__ wqkvT,
    _Float16* __restrict__ wprojT)
{
    const int NX = B_ * S_ * D_;
    const int NW = D_ * 3 * D_;
    const int NP = D_ * D_;
    int stride = gridDim.x * blockDim.x;
    for (int i = blockIdx.x * blockDim.x + threadIdx.x; i < NX; i += stride)
        x16[i] = (_Float16)x[i];
    for (int i = blockIdx.x * blockDim.x + threadIdx.x; i < NW; i += stride) {
        int k = i / (3 * D_), n = i % (3 * D_);
        wqkvT[(size_t)n * D_ + k] = (_Float16)wqkv[i];
    }
    for (int i = blockIdx.x * blockDim.x + threadIdx.x; i < NP; i += stride) {
        int k = i / D_, n = i % D_;
        wprojT[(size_t)n * D_ + k] = (_Float16)wproj[i];
    }
}

// ---------------------------------------------------------------------------
// Kernel 2: QKV GEMM (M=4096, N=2304, K=768).
// 128x128 block tile; TDM double-buffers A(128x32) and B(128x32) in LDS;
// 8 waves, each computing a 64x32 sub-tile (8 WMMA accumulators).
// Epilogue: bias, Q scale; Q,K head-major (B*H,S,64); V transposed (B*H,64,S).
// ---------------------------------------------------------------------------
__global__ void __launch_bounds__(256) qkv_gemm_kernel(
    const _Float16* __restrict__ X, const _Float16* __restrict__ WT,
    const float* __restrict__ bias,
    _Float16* __restrict__ Qb, _Float16* __restrict__ Kb,
    _Float16* __restrict__ Vt)
{
    __shared__ __align__(16) _Float16 shA[2][128 * 32];
    __shared__ __align__(16) _Float16 shB[2][128 * 32];
    const int NBN = (3 * D_) / 128;       // 18
    int tmBase = (blockIdx.x / NBN) * 128;
    int tnBase = (blockIdx.x % NBN) * 128;
    int wid = threadIdx.x >> 5, lane = threadIdx.x & 31;
    int half = lane >> 4, r = lane & 15;
    int wm = (wid >> 2) * 64;             // 0 or 64
    int wn = (wid & 3) * 32;              // 0,32,64,96

    v8f acc[4][2];
#pragma unroll
    for (int mi = 0; mi < 4; ++mi)
#pragma unroll
        for (int ni = 0; ni < 2; ++ni) acc[mi][ni] = (v8f){};

    const int NC = D_ / 32;               // 24 K-chunks
    if (wid == 0) {
        tdm_load_tile_128x32(X  + (size_t)tmBase * D_, lds_off(&shA[0][0]), D_);
        tdm_load_tile_128x32(WT + (size_t)tnBase * D_, lds_off(&shB[0][0]), D_);
        __builtin_amdgcn_s_wait_tensorcnt(0);
    }
    __syncthreads();

    for (int c = 0; c < NC; ++c) {
        int cur = c & 1;
        if (wid == 0 && c + 1 < NC) {     // prefetch next chunk via TDM
            int k0 = (c + 1) * 32;
            tdm_load_tile_128x32(X  + (size_t)tmBase * D_ + k0,
                                 lds_off(&shA[cur ^ 1][0]), D_);
            tdm_load_tile_128x32(WT + (size_t)tnBase * D_ + k0,
                                 lds_off(&shB[cur ^ 1][0]), D_);
        }
        v16h aFrag[4], bFrag[2];
#pragma unroll
        for (int mi = 0; mi < 4; ++mi)
            aFrag[mi] = load_a_frag(&shA[cur][0], 32, wm + mi * 16 + r, 0, half);
#pragma unroll
        for (int ni = 0; ni < 2; ++ni)
            bFrag[ni] = load_b_frag(&shB[cur][0], 32, wn + ni * 16 + r, 0, half);
#pragma unroll
        for (int mi = 0; mi < 4; ++mi)
#pragma unroll
            for (int ni = 0; ni < 2; ++ni)
                acc[mi][ni] = wmma16x16x32(aFrag[mi], bFrag[ni], acc[mi][ni]);

        if (wid == 0 && c + 1 < NC) __builtin_amdgcn_s_wait_tensorcnt(0);
        __syncthreads();
    }

#pragma unroll
    for (int ni = 0; ni < 2; ++ni) {
        int n = tnBase + wn + ni * 16 + r;
        int three = n / D_;
        int h  = (n % D_) / HD_;
        int hd = n % HD_;
        float bv = bias[n];
#pragma unroll
        for (int mi = 0; mi < 4; ++mi) {
#pragma unroll
            for (int v = 0; v < 8; ++v) {
                int m  = tmBase + wm + mi * 16 + v + 8 * half;
                int bi = m / S_, si = m % S_;
                float val = acc[mi][ni][v] + bv;
                if (three == 0) {
                    Qb[(((size_t)(bi * H_ + h)) * S_ + si) * HD_ + hd] =
                        (_Float16)(val * SCALE_);
                } else if (three == 1) {
                    Kb[(((size_t)(bi * H_ + h)) * S_ + si) * HD_ + hd] =
                        (_Float16)val;
                } else {
                    Vt[(((size_t)(bi * H_ + h)) * HD_ + hd) * S_ + si] =
                        (_Float16)val;
                }
            }
        }
    }
}

// ---------------------------------------------------------------------------
// Kernel 3: flash attention.  One wave = one (b,h) x 16-query tile; online
// softmax over 64 key-blocks of 32.  P tile transposed D-layout -> A-layout
// through a private 1KB LDS region per wave.
// ---------------------------------------------------------------------------
__global__ void __launch_bounds__(256) attn_kernel(
    const _Float16* __restrict__ Q, const _Float16* __restrict__ K,
    const _Float16* __restrict__ Vt, _Float16* __restrict__ AO)
{
    __shared__ __align__(16) _Float16 sh[8][16 * 32];
    const int QT = S_ / 16;
    int wid  = threadIdx.x >> 5;
    int lane = threadIdx.x & 31;
    int wave = blockIdx.x * 8 + wid;
    int bh = wave / QT;
    int q0 = (wave % QT) * 16;
    int half = lane >> 4, r = lane & 15;

    const _Float16* Qp = Q  + (size_t)bh * S_ * HD_;
    const _Float16* Kp = K  + (size_t)bh * S_ * HD_;
    const _Float16* Vp = Vt + (size_t)bh * HD_ * S_;

    v16h a0 = load_a_frag(Qp, HD_, q0 + r, 0,  half);
    v16h a1 = load_a_frag(Qp, HD_, q0 + r, 32, half);

    v8f o0 = {}, o1 = {}, o2 = {}, o3 = {};
    float mrow[8], lrow[8];
#pragma unroll
    for (int v = 0; v < 8; ++v) { mrow[v] = -1e30f; lrow[v] = 0.f; }

    _Float16* my = &sh[wid][0];

    for (int kk = 0; kk < S_; kk += 32) {
        v8f s0 = {}, s1 = {};
        s0 = wmma16x16x32(a0, load_b_frag(Kp, HD_, kk + r,      0,  half), s0);
        s0 = wmma16x16x32(a1, load_b_frag(Kp, HD_, kk + r,      32, half), s0);
        s1 = wmma16x16x32(a0, load_b_frag(Kp, HD_, kk + 16 + r, 0,  half), s1);
        s1 = wmma16x16x32(a1, load_b_frag(Kp, HD_, kk + 16 + r, 32, half), s1);

        float rmax[8];
#pragma unroll
        for (int v = 0; v < 8; ++v) rmax[v] = fmaxf(s0[v], s1[v]);
#pragma unroll
        for (int m = 1; m < 16; m <<= 1) {
#pragma unroll
            for (int v = 0; v < 8; ++v)
                rmax[v] = fmaxf(rmax[v], __shfl_xor(rmax[v], m, 32));
        }
        float alpha[8], rsum[8];
#pragma unroll
        for (int v = 0; v < 8; ++v) {
            float mn = fmaxf(mrow[v], rmax[v]);
            alpha[v] = __expf(mrow[v] - mn);
            mrow[v]  = mn;
            float p0 = __expf(s0[v] - mn);
            float p1 = __expf(s1[v] - mn);
            s0[v] = p0; s1[v] = p1;
            rsum[v] = p0 + p1;
        }
#pragma unroll
        for (int m = 1; m < 16; m <<= 1) {
#pragma unroll
            for (int v = 0; v < 8; ++v)
                rsum[v] += __shfl_xor(rsum[v], m, 32);
        }
#pragma unroll
        for (int v = 0; v < 8; ++v) {
            lrow[v] = lrow[v] * alpha[v] + rsum[v];
            o0[v] *= alpha[v]; o1[v] *= alpha[v];
            o2[v] *= alpha[v]; o3[v] *= alpha[v];
        }

#pragma unroll
        for (int v = 0; v < 8; ++v) {
            int row = v + 8 * half;
            my[row * 32 + r]      = (_Float16)s0[v];
            my[row * 32 + 16 + r] = (_Float16)s1[v];
        }
        __builtin_amdgcn_wave_barrier();
        v16h pa;
        {
            const _Float16* p = my + r * 32 + 8 * half;
            v8h lo = *(const v8h*)p;
            v8h hi = *(const v8h*)(p + 16);
            pa = __builtin_shufflevector(lo, hi,
                     0,1,2,3,4,5,6,7,8,9,10,11,12,13,14,15);
        }
        __builtin_amdgcn_wave_barrier();

        o0 = wmma16x16x32(pa, load_b_frag(Vp, S_,  0 + r, kk, half), o0);
        o1 = wmma16x16x32(pa, load_b_frag(Vp, S_, 16 + r, kk, half), o1);
        o2 = wmma16x16x32(pa, load_b_frag(Vp, S_, 32 + r, kk, half), o2);
        o3 = wmma16x16x32(pa, load_b_frag(Vp, S_, 48 + r, kk, half), o3);
    }

    int bi = bh / H_, h = bh % H_;
#pragma unroll
    for (int v = 0; v < 8; ++v) {
        float inv = 1.f / lrow[v];
        int si = q0 + v + 8 * half;
        size_t rowoff = ((size_t)(bi * S_ + si)) * D_ + h * HD_;
        AO[rowoff +  0 + r] = (_Float16)(o0[v] * inv);
        AO[rowoff + 16 + r] = (_Float16)(o1[v] * inv);
        AO[rowoff + 32 + r] = (_Float16)(o2[v] * inv);
        AO[rowoff + 48 + r] = (_Float16)(o3[v] * inv);
    }
}

// ---------------------------------------------------------------------------
// Kernel 4: projection GEMM (M=4096, N=768, K=768), same TDM/LDS scheme.
// ---------------------------------------------------------------------------
__global__ void __launch_bounds__(256) proj_gemm_kernel(
    const _Float16* __restrict__ A, const _Float16* __restrict__ WT,
    const float* __restrict__ bias, float* __restrict__ out)
{
    __shared__ __align__(16) _Float16 shA[2][128 * 32];
    __shared__ __align__(16) _Float16 shB[2][128 * 32];
    const int NBN = D_ / 128;             // 6
    int tmBase = (blockIdx.x / NBN) * 128;
    int tnBase = (blockIdx.x % NBN) * 128;
    int wid = threadIdx.x >> 5, lane = threadIdx.x & 31;
    int half = lane >> 4, r = lane & 15;
    int wm = (wid >> 2) * 64;
    int wn = (wid & 3) * 32;

    v8f acc[4][2];
#pragma unroll
    for (int mi = 0; mi < 4; ++mi)
#pragma unroll
        for (int ni = 0; ni < 2; ++ni) acc[mi][ni] = (v8f){};

    const int NC = D_ / 32;
    if (wid == 0) {
        tdm_load_tile_128x32(A  + (size_t)tmBase * D_, lds_off(&shA[0][0]), D_);
        tdm_load_tile_128x32(WT + (size_t)tnBase * D_, lds_off(&shB[0][0]), D_);
        __builtin_amdgcn_s_wait_tensorcnt(0);
    }
    __syncthreads();

    for (int c = 0; c < NC; ++c) {
        int cur = c & 1;
        if (wid == 0 && c + 1 < NC) {
            int k0 = (c + 1) * 32;
            tdm_load_tile_128x32(A  + (size_t)tmBase * D_ + k0,
                                 lds_off(&shA[cur ^ 1][0]), D_);
            tdm_load_tile_128x32(WT + (size_t)tnBase * D_ + k0,
                                 lds_off(&shB[cur ^ 1][0]), D_);
        }
        v16h aFrag[4], bFrag[2];
#pragma unroll
        for (int mi = 0; mi < 4; ++mi)
            aFrag[mi] = load_a_frag(&shA[cur][0], 32, wm + mi * 16 + r, 0, half);
#pragma unroll
        for (int ni = 0; ni < 2; ++ni)
            bFrag[ni] = load_b_frag(&shB[cur][0], 32, wn + ni * 16 + r, 0, half);
#pragma unroll
        for (int mi = 0; mi < 4; ++mi)
#pragma unroll
            for (int ni = 0; ni < 2; ++ni)
                acc[mi][ni] = wmma16x16x32(aFrag[mi], bFrag[ni], acc[mi][ni]);

        if (wid == 0 && c + 1 < NC) __builtin_amdgcn_s_wait_tensorcnt(0);
        __syncthreads();
    }

#pragma unroll
    for (int ni = 0; ni < 2; ++ni) {
        int n = tnBase + wn + ni * 16 + r;
        float bv = bias[n];
#pragma unroll
        for (int mi = 0; mi < 4; ++mi) {
#pragma unroll
            for (int v = 0; v < 8; ++v) {
                int m = tmBase + wm + mi * 16 + v + 8 * half;
                out[(size_t)m * D_ + n] = acc[mi][ni][v] + bv;
            }
        }
    }
}

// ---------------------------------------------------------------------------
// Host-side launch
// ---------------------------------------------------------------------------
extern "C" void kernel_launch(void* const* d_in, const int* in_sizes, int n_in,
                              void* d_out, int out_size, void* d_ws, size_t ws_size,
                              hipStream_t stream) {
    const float* x      = (const float*)d_in[0];
    const float* w_qkv  = (const float*)d_in[1];
    const float* b_qkv  = (const float*)d_in[2];
    const float* w_proj = (const float*)d_in[3];
    const float* b_proj = (const float*)d_in[4];
    float* out = (float*)d_out;

    char* ws = (char*)d_ws;
    size_t off = 0;
    auto carve = [&](size_t bytes) {
        void* p = ws + off;
        off += (bytes + 255) & ~(size_t)255;
        return p;
    };
    _Float16* x16    = (_Float16*)carve((size_t)B_ * S_ * D_ * 2);
    _Float16* wqkvT  = (_Float16*)carve((size_t)3 * D_ * D_ * 2);
    _Float16* wprojT = (_Float16*)carve((size_t)D_ * D_ * 2);
    _Float16* Qb     = (_Float16*)carve((size_t)B_ * H_ * S_ * HD_ * 2);
    _Float16* Kb     = (_Float16*)carve((size_t)B_ * H_ * S_ * HD_ * 2);
    _Float16* Vt     = (_Float16*)carve((size_t)B_ * H_ * S_ * HD_ * 2);
    _Float16* AO     = (_Float16*)carve((size_t)B_ * S_ * D_ * 2);
    (void)ws_size; (void)in_sizes; (void)n_in; (void)out_size;

    convert_kernel<<<2048, 256, 0, stream>>>(x, w_qkv, w_proj, x16, wqkvT, wprojT);

    // 32 x 18 block tiles of 128x128
    qkv_gemm_kernel<<<32 * 18, 256, 0, stream>>>(x16, wqkvT, b_qkv, Qb, Kb, Vt);

    // B*H * (S/16) = 3072 waves, 8 per block
    attn_kernel<<<3072 / 8, 256, 0, stream>>>(Qb, Kb, Vt, AO);

    // 32 x 6 block tiles of 128x128
    proj_gemm_kernel<<<32 * 6, 256, 0, stream>>>(AO, wprojT, b_proj, out);
}
